// Head_53051436040512
// MI455X (gfx1250) — matrix operand order repaired and checked by reading, not compiled, not run on hardware
//
#include <hip/hip_runtime.h>

typedef __attribute__((ext_vector_type(16))) __bf16 v16bf;
typedef __attribute__((ext_vector_type(8)))  __bf16 v8bf;
typedef __attribute__((ext_vector_type(8)))  float  v8f;

#define N_EMBD 1024
#define HEAD   64
#define BATCH  16
#define SEQ    2048
#define MTOT   (BATCH*SEQ)   /* 32768 */

// workspace layout (bytes)
#define WS_WT    0                       // bf16 [192][1024]  (W^T, fused Q|K|V)
#define WS_BIAS  (192*1024*2)            // f32  [192]
#define WS_Q     (512*1024)              // bf16 [MTOT][64]   (pre-scaled by 1/8)
#define WS_K     (WS_Q  + MTOT*HEAD*2)   // bf16 [MTOT][64]
#define WS_VT    (WS_K  + MTOT*HEAD*2)   // bf16 [64][MTOT]   (V transposed)

__device__ __forceinline__ v8f zero8() {
  v8f z;
#pragma unroll
  for (int i = 0; i < 8; i++) z[i] = 0.0f;
  return z;
}

// LDS byte offset of a generic pointer to a __shared__ object.
// (AS3 pointers are 32-bit on the device; host pass must not see the cast.)
__device__ __forceinline__ unsigned int lds_off(void* p) {
#ifdef __HIP_DEVICE_COMPILE__
  return (unsigned int)(unsigned long long)(__attribute__((address_space(3))) void*)p;
#else
  return 0u;
#endif
}

// CDNA5 async copy: 16 bytes global -> LDS, tracked with ASYNCcnt
__device__ __forceinline__ void async_ld_b128(unsigned int lds_dst, const void* gsrc) {
  asm volatile("global_load_async_to_lds_b128 %0, %1, off"
               :: "v"(lds_dst), "v"((unsigned long long)gsrc)
               : "memory");
}

// ---------------------------------------------------------------------------
// Kernel 0: convert weights to bf16, transposed: WT[n][k] = W[k][n], n in [0,192)
// ---------------------------------------------------------------------------
__global__ void pack_weights(const float* __restrict__ Wq, const float* __restrict__ bq,
                             const float* __restrict__ Wk, const float* __restrict__ bk,
                             const float* __restrict__ Wv, const float* __restrict__ bv,
                             __bf16* __restrict__ WT, float* __restrict__ bias) {
  int tid = blockIdx.x * 256 + threadIdx.x;
  if (tid < 192 * 1024) {
    int n = tid >> 10, k = tid & 1023;
    const float* W = (n < 64) ? Wq : (n < 128) ? Wk : Wv;
    WT[tid] = (__bf16)W[k * HEAD + (n & 63)];
  }
  if (tid < 192) {
    const float* bb = (tid < 64) ? bq : (tid < 128) ? bk : bv;
    bias[tid] = bb[tid & 63];
  }
}

// ---------------------------------------------------------------------------
// Kernel 1: QKV projection. Block = 256 threads (8 waves), 128 rows/block.
// The W^T k-slice (192x32 bf16 = 12 KB) is async-staged into a double-
// buffered LDS tile one iteration ahead; B fragments are ds_load_b128.
// ---------------------------------------------------------------------------
__global__ void __launch_bounds__(256)
qkv_proj(const float* __restrict__ x, const __bf16* __restrict__ WT,
         const float* __restrict__ bias,
         __bf16* __restrict__ Q, __bf16* __restrict__ K, __bf16* __restrict__ VT) {
  __shared__ __bf16 wbuf[2 * 192 * 32];  // double-buffered W^T slice, 24 KB

  const int lane = threadIdx.x & 31;
  const int wave = threadIdx.x >> 5;
  const int col  = lane & 15;
  const int hf   = lane >> 4;
  const int m0   = blockIdx.x * 128 + wave * 16;

  // stage one 192x32 W^T slice: 768 x 16B chunks, 3 async b128 per thread
  auto stageW = [&](int kt, int sel) {
#pragma unroll
    for (int i = 0; i < 3; i++) {
      int c   = threadIdx.x + i * 256;
      int row = c >> 2, sub = c & 3;     // row 0..191, 4 chunks of 8 bf16
      async_ld_b128(lds_off(&wbuf[sel * 6144 + row * 32 + sub * 8]),
                    WT + (size_t)row * N_EMBD + kt * 32 + sub * 8);
    }
  };

  v8f acc[12];
#pragma unroll
  for (int i = 0; i < 12; i++) acc[i] = zero8();

  const float* xrow = x + (size_t)(m0 + col) * N_EMBD;

  stageW(0, 0);                           // prologue: slice 0 -> buffer 0

  for (int kt = 0; kt < N_EMBD / 32; kt++) {
    if (kt < N_EMBD / 32 - 1) {
      stageW(kt + 1, (kt + 1) & 1);       // prefetch next slice
      asm volatile("s_wait_asynccnt 0x3" ::: "memory");  // slice kt resident
    } else {
      asm volatile("s_wait_asynccnt 0x0" ::: "memory");
    }
    __syncthreads();

    // A fragment: convert fp32 -> bf16 on the fly
    const float* pa = xrow + kt * 32 + hf * 16;
    union { v16bf v; __bf16 e[16]; } A;
#pragma unroll
    for (int i = 0; i < 8; i++) A.e[i]     = (__bf16)pa[i];
#pragma unroll
    for (int i = 0; i < 8; i++) A.e[8 + i] = (__bf16)pa[8 + i];

    const __bf16* wb = wbuf + (kt & 1) * 6144;
#pragma unroll
    for (int nt = 0; nt < 12; nt++) {
      v16bf B = *(const v16bf*)(wb + (nt * 16 + col) * 32 + hf * 16);
      acc[nt] = __builtin_amdgcn_wmma_f32_16x16x32_bf16(false, A.v, false, B,
                                                        (short)0, acc[nt], false, false);
    }
    __syncthreads();   // all waves done reading before buffer is overwritten
  }

  // bias
#pragma unroll
  for (int nt = 0; nt < 12; nt++) {
    float bb = bias[nt * 16 + col];
#pragma unroll
    for (int r = 0; r < 8; r++) acc[nt][r] += bb;
  }
  // Q (fold softmax scale 1/sqrt(64) = 0.125)
#pragma unroll
  for (int nt = 0; nt < 4; nt++)
#pragma unroll
    for (int r = 0; r < 8; r++) {
      int m = m0 + r + 8 * hf;
      Q[(size_t)m * HEAD + nt * 16 + col] = (__bf16)(acc[nt][r] * 0.125f);
    }
  // K
#pragma unroll
  for (int nt = 4; nt < 8; nt++)
#pragma unroll
    for (int r = 0; r < 8; r++) {
      int m = m0 + r + 8 * hf;
      K[(size_t)m * HEAD + (nt - 4) * 16 + col] = (__bf16)acc[nt][r];
    }
  // V transposed: lane holds 8 consecutive m for fixed h -> single 16B store
#pragma unroll
  for (int nt = 8; nt < 12; nt++) {
    union { v8bf v; __bf16 e[8]; } V8;
#pragma unroll
    for (int r = 0; r < 8; r++) V8.e[r] = (__bf16)acc[nt][r];
    int h = (nt - 8) * 16 + col;
    *(v8bf*)(VT + (size_t)h * MTOT + m0 + 8 * hf) = V8.v;
  }
}

// ---------------------------------------------------------------------------
// Kernel 2: flash attention, block-cooperative.
// grid = (BATCH, SEQ/128), block = 256 (8 waves); all waves share async
// double-buffered K/V tiles in LDS and run a block-uniform tile count
// (fully-masked tiles yield p=0, alpha=1 -> numerically identical).
// ---------------------------------------------------------------------------
__global__ void __launch_bounds__(256)
attention(const __bf16* __restrict__ Q, const __bf16* __restrict__ K,
          const __bf16* __restrict__ VT, float* __restrict__ out) {
  __shared__ __bf16 kbuf[2 * 32 * 64];   // [buf][key 0..31][head 0..63]   8 KB
  __shared__ __bf16 vbuf[2 * 64 * 32];   // [buf][head 0..63][key 0..31]   8 KB
  __shared__ __bf16 plds[8 * 16 * 32];   // per-wave P staging              8 KB

  const int tid  = threadIdx.x;
  const int lane = tid & 31;
  const int wave = tid >> 5;
  const int col  = lane & 15;
  const int hf   = lane >> 4;
  const int b    = blockIdx.x;
  const int qb   = blockIdx.y * 128;     // block's first query row
  const int q0   = qb + wave * 16;       // wave's first query row
  const size_t base = (size_t)b * SEQ;

  // stage one 32-key tile (K: 4KB, V^T: 4KB): 2 async b128 per thread
  auto stage = [&](int kt, int sel) {
    int sK = tid >> 3, cK = tid & 7;     // 32 key rows x 8 chunks of 8 bf16
    async_ld_b128(lds_off(&kbuf[sel * 2048 + sK * 64 + cK * 8]),
                  K + (base + (size_t)kt * 32 + sK) * HEAD + cK * 8);
    int hV = tid >> 2, cV = tid & 3;     // 64 head rows x 4 chunks of 8 bf16
    async_ld_b128(lds_off(&vbuf[sel * 2048 + hV * 32 + cV * 8]),
                  VT + (size_t)hV * MTOT + base + (size_t)kt * 32 + cV * 8);
  };

  // Q A-fragments (two 16x32 slices covering head dim 0..63), pre-scaled
  v16bf qa[2];
  {
    const __bf16* qrow = Q + (base + q0 + col) * HEAD;
#pragma unroll
    for (int kf = 0; kf < 2; kf++) {
      union { v16bf v; v8bf h[2]; } u;
      u.h[0] = *(const v8bf*)(qrow + kf * 32 + hf * 16);
      u.h[1] = *(const v8bf*)(qrow + kf * 32 + hf * 16 + 8);
      qa[kf] = u.v;
    }
  }

  v8f o[4];
#pragma unroll
  for (int i = 0; i < 4; i++) o[i] = zero8();
  float mrun[8], lrun[8];
#pragma unroll
  for (int r = 0; r < 8; r++) { mrun[r] = -1e30f; lrun[r] = 0.0f; }

  __bf16* pw = plds + wave * 512;
  const int ktMax = (qb + 127) >> 5;     // block-uniform causal frontier

  stage(0, 0);                            // prologue: tile 0 -> buffer 0

  for (int kt = 0; kt <= ktMax; kt++) {
    if (kt < ktMax) {
      stage(kt + 1, (kt + 1) & 1);        // prefetch next tile
      asm volatile("s_wait_asynccnt 0x2" ::: "memory");  // tile kt resident
    } else {
      asm volatile("s_wait_asynccnt 0x0" ::: "memory");
    }
    __syncthreads();

    const __bf16* kb = kbuf + (kt & 1) * 2048;
    const __bf16* vb = vbuf + (kt & 1) * 2048;
    const int s0base = kt * 32;

    // S = Q @ K^T for a 16x32 tile (fragments from LDS)
    v8f s[2];
#pragma unroll
    for (int sub = 0; sub < 2; sub++) {
      const __bf16* kr = kb + (sub * 16 + col) * 64 + hf * 16;
      v16bf kb0 = *(const v16bf*)(kr);
      v16bf kb1 = *(const v16bf*)(kr + 32);
      v8f cz = zero8();
      cz     = __builtin_amdgcn_wmma_f32_16x16x32_bf16(false, qa[0], false, kb0,
                                                       (short)0, cz, false, false);
      s[sub] = __builtin_amdgcn_wmma_f32_16x16x32_bf16(false, qa[1], false, kb1,
                                                       (short)0, cz, false, false);
    }

    // causal mask (C/D layout: lane holds rows r+8*hf, column col)
#pragma unroll
    for (int sub = 0; sub < 2; sub++) {
      int n = s0base + sub * 16 + col;
#pragma unroll
      for (int r = 0; r < 8; r++) {
        int m = q0 + r + 8 * hf;
        if (n > m) s[sub][r] = -1e30f;
      }
    }

    // online softmax: row stats live within each 16-lane half (xor masks 1..8)
    float p0[8], p1[8], alpha[8];
#pragma unroll
    for (int r = 0; r < 8; r++) {
      float vm = fmaxf(s[0][r], s[1][r]);
#pragma unroll
      for (int off = 1; off < 16; off <<= 1)
        vm = fmaxf(vm, __shfl_xor(vm, off, 32));
      float nm = fmaxf(mrun[r], vm);
      alpha[r] = __expf(mrun[r] - nm);
      mrun[r]  = nm;
      p0[r] = __expf(s[0][r] - nm);
      p1[r] = __expf(s[1][r] - nm);
      float ps = p0[r] + p1[r];
#pragma unroll
      for (int off = 1; off < 16; off <<= 1)
        ps += __shfl_xor(ps, off, 32);
      lrun[r] = lrun[r] * alpha[r] + ps;
    }

    // stage P (D-layout -> row-major 16x32 bf16 in per-wave LDS region)
#pragma unroll
    for (int r = 0; r < 8; r++) {
      int m = r + 8 * hf;
      pw[m * 32 + col]      = (__bf16)p0[r];
      pw[m * 32 + 16 + col] = (__bf16)p1[r];
    }
    asm volatile("s_wait_dscnt 0x0" ::: "memory");

    // reload P in A-fragment layout (lane m = col)
    union { v16bf v; v8bf h[2]; } pu;
    pu.h[0] = *(const v8bf*)(pw + col * 32 + hf * 16);
    pu.h[1] = *(const v8bf*)(pw + col * 32 + hf * 16 + 8);

    // O = O*alpha + P @ V   (V^T tile in LDS -> contiguous B-fragments)
#pragma unroll
    for (int nf = 0; nf < 4; nf++) {
#pragma unroll
      for (int r = 0; r < 8; r++) o[nf][r] *= alpha[r];
      v16bf vfrag = *(const v16bf*)(vb + (nf * 16 + col) * 32 + hf * 16);
      o[nf] = __builtin_amdgcn_wmma_f32_16x16x32_bf16(false, pu.v, false, vfrag,
                                                      (short)0, o[nf], false, false);
    }
    __syncthreads();   // everyone done reading this buffer before overwrite
  }

  // normalize and store fp32 output [B, T, 64]
#pragma unroll
  for (int nf = 0; nf < 4; nf++)
#pragma unroll
    for (int r = 0; r < 8; r++) {
      int m = q0 + r + 8 * hf;
      out[(base + m) * HEAD + nf * 16 + col] = o[nf][r] / lrun[r];
    }
}

// ---------------------------------------------------------------------------
extern "C" void kernel_launch(void* const* d_in, const int* in_sizes, int n_in,
                              void* d_out, int out_size, void* d_ws, size_t ws_size,
                              hipStream_t stream) {
  (void)in_sizes; (void)n_in; (void)out_size; (void)ws_size;
  const float* x  = (const float*)d_in[0];
  const float* Wq = (const float*)d_in[1];
  const float* bq = (const float*)d_in[2];
  const float* Wk = (const float*)d_in[3];
  const float* bk = (const float*)d_in[4];
  const float* Wv = (const float*)d_in[5];
  const float* bv = (const float*)d_in[6];

  char* ws = (char*)d_ws;
  __bf16* WT  = (__bf16*)(ws + WS_WT);
  float*  bia = (float*)(ws + WS_BIAS);
  __bf16* Q   = (__bf16*)(ws + WS_Q);
  __bf16* K   = (__bf16*)(ws + WS_K);
  __bf16* VT  = (__bf16*)(ws + WS_VT);

  pack_weights<<<(192 * 1024 + 255) / 256, 256, 0, stream>>>(Wq, bq, Wk, bk, Wv, bv, WT, bia);
  qkv_proj<<<MTOT / 128, 256, 0, stream>>>(x, WT, bia, Q, K, VT);
  dim3 g(BATCH, SEQ / 128);
  attention<<<g, 256, 0, stream>>>(Q, K, VT, (float*)d_out);
}
